// AutoregressiveLSTM_3624952398025
// MI455X (gfx1250) — matrix operand order repaired and checked by reading, not compiled, not run on hardware
//
#include <hip/hip_runtime.h>

typedef __attribute__((ext_vector_type(16))) _Float16 v16h;
typedef __attribute__((ext_vector_type(8)))  float    v8f;
typedef __attribute__((ext_vector_type(8)))  unsigned v8u;

#define BATCH   8192
#define SEQ     200
#define INP     4
#define HID     10
#define OUTD    4
#define PRED    5
#define TPRIME  (SEQ - PRED + 1)   /* 196 */

#define NLOG2E  (-1.44269504f)     /* -log2(e)   : sigmoid scale */
#define N2LOG2E (-2.88539008f)     /* -2*log2(e) : tanh scale    */

// pack two f32 -> one dword of two f16
static __device__ __forceinline__ unsigned pkh(float a, float b) {
  union { _Float16 h[2]; unsigned u; } t;
  t.h[0] = (_Float16)a; t.h[1] = (_Float16)b;
  return t.u;
}

// exchange lane l <-> lane l^16 (ds_swizzle SWAPX16: xor=0x10, and=0x1f)
static __device__ __forceinline__ float swap16(float x) {
  int r = __builtin_amdgcn_ds_swizzle(__builtin_bit_cast(int, x), 0x401F);
  return __builtin_bit_cast(float, r);
}

// sigmoid(x + b) with pre-scaled bias nb = NLOG2E * b:  rcp(1 + exp2(fma(x,NLOG2E,nb)))
static __device__ __forceinline__ float sigm_b(float x, float nb) {
  return __builtin_amdgcn_rcpf(1.0f + __builtin_amdgcn_exp2f(__builtin_fmaf(x, NLOG2E, nb)));
}
// tanh(x + b) with pre-scaled bias nb2 = N2LOG2E * b
static __device__ __forceinline__ float tanh_b(float x, float nb2) {
  return 2.0f * __builtin_amdgcn_rcpf(1.0f + __builtin_amdgcn_exp2f(__builtin_fmaf(x, N2LOG2E, nb2))) - 1.0f;
}
// plain tanh (for cell state)
static __device__ __forceinline__ float tanh_(float x) {
  return 2.0f * __builtin_amdgcn_rcpf(1.0f + __builtin_amdgcn_exp2f(x * N2LOG2E)) - 1.0f;
}

// D = A(16x32 f16) * B(32x16 f16) + 0   (zero C folds to inline SRC2=0 -> no copy-moves)
static __device__ __forceinline__ v8f wmma16z(v16h A, v16h B) {
  const v8f z = {0, 0, 0, 0, 0, 0, 0, 0};
  return __builtin_amdgcn_wmma_f32_16x16x32_f16(false, A, false, B, (short)0, z, false, false);
}

// B-matrix (32x16 f16): column n = batch lane (0..15), rows K = [x0..x3, h0..h9, 0, 0].
// Lanes 16..31 carry K=16..31 -> all zero. h8,h9 live in the upper lane half of the
// accumulator-layout h, fetched with one SWAPX16 each.
static __device__ __forceinline__ v16h buildB_cell(float i0, float i1, float i2, float i3,
                                                   v8f h, bool lo) {
  float s0 = swap16(h[0]);
  float s1 = swap16(h[1]);
  v8u u;
  u[0] = lo ? pkh(i0, i1)     : 0u;
  u[1] = lo ? pkh(i2, i3)     : 0u;
  u[2] = lo ? pkh(h[0], h[1]) : 0u;
  u[3] = lo ? pkh(h[2], h[3]) : 0u;
  u[4] = lo ? pkh(h[4], h[5]) : 0u;
  u[5] = lo ? pkh(h[6], h[7]) : 0u;
  u[6] = lo ? pkh(s0, s1)     : 0u;
  u[7] = 0u;
  return __builtin_bit_cast(v16h, u);
}

// B-matrix for the linear head: rows K = [h0..h9, 0..]
static __device__ __forceinline__ v16h buildB_lin(v8f h, bool lo) {
  float s0 = swap16(h[0]);
  float s1 = swap16(h[1]);
  v8u u;
  u[0] = lo ? pkh(h[0], h[1]) : 0u;
  u[1] = lo ? pkh(h[2], h[3]) : 0u;
  u[2] = lo ? pkh(h[4], h[5]) : 0u;
  u[3] = lo ? pkh(h[6], h[7]) : 0u;
  u[4] = lo ? pkh(s0, s1)     : 0u;
  u[5] = 0u; u[6] = 0u; u[7] = 0u;
  return __builtin_bit_cast(v16h, u);
}

__global__ __launch_bounds__(128)
void lstm_ar_wmma(const float* __restrict__ x,
                  const float* __restrict__ Wih, const float* __restrict__ Whh,
                  const float* __restrict__ bih, const float* __restrict__ bhh,
                  const float* __restrict__ Wlin, const float* __restrict__ blin,
                  float* __restrict__ outF, float* __restrict__ outC)
{
  const int  lane = (int)(threadIdx.x & 31u);
  const bool lo   = lane < 16;
  const int  row  = lane & 15;                       // A-matrix row / batch column
  const int  wave = (int)((blockIdx.x * blockDim.x + threadIdx.x) >> 5);
  const int  b    = wave * 16 + row;                 // this lane's batch element

  // ---- constant A tiles in VGPRs: 4 gate tiles (16 rows, 10 used; K=14 of 32) + head ----
  // A packed here in contiguous k order (as B): lanes 0-15 row=lane K=0..15 (v0-7);
  //                     lanes 16-31 row=lane-16 K=16..31.
  v16h WA[4], WL;
  #pragma unroll
  for (int g = 0; g < 4; ++g) {
    v8u u;
    #pragma unroll
    for (int v = 0; v < 8; ++v) {
      int k0 = (lo ? 0 : 16) + 2 * v;
      float a = 0.f, c2 = 0.f;
      if (row < HID) {
        int gr = g * HID + row;                      // PyTorch gate order i,f,g,o
        int ka = k0, kb = k0 + 1;
        if (ka < INP) a  = Wih[gr * INP + ka]; else if (ka < INP + HID) a  = Whh[gr * HID + (ka - INP)];
        if (kb < INP) c2 = Wih[gr * INP + kb]; else if (kb < INP + HID) c2 = Whh[gr * HID + (kb - INP)];
      }
      u[v] = pkh(a, c2);
    }
    WA[g] = __builtin_bit_cast(v16h, u);
  }
  {
    v8u u;
    #pragma unroll
    for (int v = 0; v < 8; ++v) {
      int k0 = (lo ? 0 : 16) + 2 * v;
      float a = 0.f, c2 = 0.f;
      if (row < OUTD) {
        if (k0     < HID) a  = Wlin[row * HID + k0];
        if (k0 + 1 < HID) c2 = Wlin[row * HID + k0 + 1];
      }
      u[v] = pkh(a, c2);
    }
    WL = __builtin_bit_cast(v16h, u);
  }

  // ---- biases pre-scaled into the activation fma constants (C operand stays 0) ----
  // gates i,f,o consumed via sigmoid -> scale NLOG2E; gate g via tanh -> scale N2LOG2E
  v8f NBi, NBf, NBg, NBo;
  #pragma unroll
  for (int m = 0; m < 8; ++m) {
    int r = m + (lo ? 0 : 8);
    float bi = (r < HID) ? (bih[0 * HID + r] + bhh[0 * HID + r]) : 0.f;
    float bf = (r < HID) ? (bih[1 * HID + r] + bhh[1 * HID + r]) : 0.f;
    float bg = (r < HID) ? (bih[2 * HID + r] + bhh[2 * HID + r]) : 0.f;
    float bo = (r < HID) ? (bih[3 * HID + r] + bhh[3 * HID + r]) : 0.f;
    NBi[m] = NLOG2E  * bi;
    NBf[m] = NLOG2E  * bf;
    NBg[m] = N2LOG2E * bg;
    NBo[m] = NLOG2E  * bo;
  }
  // head bias: only rows 0..3 of the lower lane-half carry valid outputs
  float bl0 = lo ? blin[0] : 0.f;
  float bl1 = lo ? blin[1] : 0.f;
  float bl2 = lo ? blin[2] : 0.f;
  float bl3 = lo ? blin[3] : 0.f;

  v8f h = {0, 0, 0, 0, 0, 0, 0, 0};
  v8f c = h;

  for (int t = 0; t < TPRIME; ++t) {
    const float4 xt = *reinterpret_cast<const float4*>(x + ((size_t)b * SEQ + t) * INP);
    if (t + 1 < TPRIME)
      __builtin_prefetch(x + ((size_t)b * SEQ + t + 1) * INP, 0, 0);   // global_prefetch

    // ---- teacher-forced cell ----
    {
      v16h Bm = buildB_cell(xt.x, xt.y, xt.z, xt.w, h, lo);
      v8f gi = wmma16z(WA[0], Bm);
      v8f gf = wmma16z(WA[1], Bm);
      v8f gg = wmma16z(WA[2], Bm);
      v8f go = wmma16z(WA[3], Bm);
      #pragma unroll
      for (int m = 0; m < 8; ++m) {
        c[m] = sigm_b(gf[m], NBf[m]) * c[m] + sigm_b(gi[m], NBi[m]) * tanh_b(gg[m], NBg[m]);
        h[m] = sigm_b(go[m], NBo[m]) * tanh_(c[m]);
      }
    }
    v8f y = wmma16z(WL, buildB_lin(h, lo));
    float y0 = y[0] + bl0, y1 = y[1] + bl1, y2 = y[2] + bl2, y3 = y[3] + bl3;

    float* op = outF + (((size_t)b * TPRIME + t) * PRED) * OUTD;
    if (lo) *reinterpret_cast<float4*>(op) = make_float4(y0, y1, y2, y3);

    // ---- autoregressive rollout (4 extra cells on a branched state) ----
    v8f ha = h, ca = c;
    #pragma unroll
    for (int s = 1; s < PRED; ++s) {
      v16h Bm = buildB_cell(y0, y1, y2, y3, ha, lo);
      v8f gi = wmma16z(WA[0], Bm);
      v8f gf = wmma16z(WA[1], Bm);
      v8f gg = wmma16z(WA[2], Bm);
      v8f go = wmma16z(WA[3], Bm);
      #pragma unroll
      for (int m = 0; m < 8; ++m) {
        ca[m] = sigm_b(gf[m], NBf[m]) * ca[m] + sigm_b(gi[m], NBi[m]) * tanh_b(gg[m], NBg[m]);
        ha[m] = sigm_b(go[m], NBo[m]) * tanh_(ca[m]);
      }
      v8f ys = wmma16z(WL, buildB_lin(ha, lo));
      y0 = ys[0] + bl0; y1 = ys[1] + bl1; y2 = ys[2] + bl2; y3 = ys[3] + bl3;
      if (lo) *reinterpret_cast<float4*>(op + s * OUTD) = make_float4(y0, y1, y2, y3);
    }
  }

  // ---- final cell state c_t: [1, B, HID]; rows 0-7 in lower lanes, 8-9 in upper ----
  float* cp = outC + (size_t)b * HID;
  if (lo) {
    *reinterpret_cast<float2*>(cp + 0) = make_float2(c[0], c[1]);
    *reinterpret_cast<float2*>(cp + 2) = make_float2(c[2], c[3]);
    *reinterpret_cast<float2*>(cp + 4) = make_float2(c[4], c[5]);
    *reinterpret_cast<float2*>(cp + 6) = make_float2(c[6], c[7]);
  } else {
    *reinterpret_cast<float2*>(cp + 8) = make_float2(c[0], c[1]);
  }
}

extern "C" void kernel_launch(void* const* d_in, const int* in_sizes, int n_in,
                              void* d_out, int out_size, void* d_ws, size_t ws_size,
                              hipStream_t stream) {
  (void)in_sizes; (void)n_in; (void)out_size; (void)d_ws; (void)ws_size;
  const float* x    = (const float*)d_in[0];
  const float* Wih  = (const float*)d_in[1];
  const float* Whh  = (const float*)d_in[2];
  const float* bih  = (const float*)d_in[3];
  const float* bhh  = (const float*)d_in[4];
  const float* Wlin = (const float*)d_in[5];
  const float* blin = (const float*)d_in[6];
  float* outF = (float*)d_out;
  float* outC = outF + (size_t)BATCH * TPRIME * PRED * OUTD;

  dim3 block(128);                       // 4 waves / block, 16 batch elems / wave
  dim3 grid(BATCH / (16 * 4));           // 128 blocks -> 512 waves total
  hipLaunchKernelGGL(lstm_ar_wmma, grid, block, 0, stream,
                     x, Wih, Whh, bih, bhh, Wlin, blin, outF, outC);
}
